// IMEAttention_16647293239319
// MI455X (gfx1250) — matrix operand order repaired and checked by reading, not compiled, not run on hardware
//
#include <hip/hip_runtime.h>

typedef _Float16 half_t;
typedef __attribute__((ext_vector_type(16))) _Float16 v16h;
typedef __attribute__((ext_vector_type(8)))  _Float16 v8h;
typedef __attribute__((ext_vector_type(8)))  float    v8f;

#define WMMA_F16(A, B, C) \
    __builtin_amdgcn_wmma_f32_16x16x32_f16(false, (A), false, (B), (short)0, (C), false, false)

// Combine two aligned 8-half LDS reads into one 16-half WMMA fragment.
static __device__ __forceinline__ v16h load_frag(const half_t* lo, const half_t* hi) {
    v8h a = *(const v8h*)lo;
    v8h b = *(const v8h*)hi;
    return __builtin_shufflevector(a, b, 0,1,2,3,4,5,6,7,8,9,10,11,12,13,14,15);
}

// CDNA5 async copy: 16 bytes global -> LDS, tracked with ASYNCcnt (no VGPR data).
static __device__ __forceinline__ void async_copy_b128(const half_t* gsrc, half_t* ldst) {
    unsigned lds_off = (unsigned)(uintptr_t)ldst;  // LDS aperture: low 32 bits = LDS address
    asm volatile("global_load_async_to_lds_b128 %0, %1, off"
                 :: "v"(lds_off), "v"(gsrc) : "memory");
}
static __device__ __forceinline__ void wait_async() {
    asm volatile("s_wait_asynccnt 0" ::: "memory");
}

// Butterfly max-reduction across the 16 lanes of each half-wave.
// Prefer v_permlane16_b32 (VALU crossbar) over ds_bpermute (LDS unit).
#if __has_builtin(__builtin_amdgcn_permlane16)
static __device__ __forceinline__ float pl16(float v, unsigned s0, unsigned s1) {
    unsigned u = __builtin_bit_cast(unsigned, v);
    unsigned r = __builtin_amdgcn_permlane16(u, u, s0, s1, false, false);
    return __builtin_bit_cast(float, r);
}
static __device__ __forceinline__ float reduce_max16(float v) {
    v = fmaxf(v, pl16(v, 0x67452301u, 0xEFCDAB89u));  // xor 1
    v = fmaxf(v, pl16(v, 0x54761032u, 0xDCFE98BAu));  // xor 2
    v = fmaxf(v, pl16(v, 0x32107654u, 0xBA98FEDCu));  // xor 4
    v = fmaxf(v, pl16(v, 0xFEDCBA98u, 0x76543210u));  // xor 8
    return v;
}
#else
static __device__ __forceinline__ float reduce_max16(float v) {
    for (int g = 1; g < 16; g <<= 1) v = fmaxf(v, __shfl_xor(v, g, 32));
    return v;
}
#endif

// ---------------------------------------------------------------------------
// Generic GEMM + bias:  Out[M,N] = A[M,K] * W[K,N] + bias[N]
// 128x128 workgroup tile, K-step 32, 8 waves (4Mx2N), 8 WMMA/wave/step.
// Double-buffered LDS: stage tile i+1 (async DMA for f16 A) while computing i.
// ---------------------------------------------------------------------------
template<bool A_F16, bool OUT_F32>
__global__ __launch_bounds__(256) void gemm_bias_kernel(
    const void* __restrict__ Ain, const float* __restrict__ W,
    const float* __restrict__ bias, void* __restrict__ Out,
    int M, int N, int K)
{
    __shared__ __align__(16) half_t sA[2][128 * 32];   // [m][k]
    __shared__ __align__(16) half_t sB[2][128 * 32];   // [n][k] (transposed stage)

    const int tid   = threadIdx.x;
    const int wave  = tid >> 5;
    const int lane  = tid & 31;
    const int lh    = lane >> 4;     // half-wave id (0/1)
    const int ll    = lane & 15;     // lane within half
    const int waveM = wave & 3;      // 4 M-groups of 32 rows
    const int waveN = wave >> 2;     // 2 N-groups of 64 cols
    const int mb    = blockIdx.y * 128;
    const int nb    = blockIdx.x * 128;

    auto stage = [&](int buf, int kb) {
        half_t* dA = sA[buf];
        half_t* dB = sB[buf];
        if (A_F16) {
            // Raw f16 tile: async DMA to LDS, 2x16B per thread.
#pragma unroll
            for (int j = 0; j < 2; ++j) {
                const int chunk = tid + j * 256;        // 512 x 8-half chunks
                const int r = chunk >> 2;               // 4 chunks per 32-half row
                const int k = (chunk & 3) * 8;
                async_copy_b128(((const half_t*)Ain) + (size_t)(mb + r) * K + kb + k,
                                &dA[r * 32 + k]);
            }
        } else {
            for (int i = tid; i < 128 * 32; i += 256) {
                const int r = i >> 5, k = i & 31;
                dA[i] = (half_t)((const float*)Ain)[(size_t)(mb + r) * K + kb + k];
            }
        }
        // W tile transposed: dB[n][k] = W[kb+k][nb+n] (fp32 -> f16).
        for (int i = tid; i < 128 * 32; i += 256) {
            const int k = i >> 7, n = i & 127;
            dB[n * 32 + k] = (half_t)W[(size_t)(kb + k) * N + nb + n];
        }
        // Prefetch the K-slab after next (global_prefetch_b8).
        if (kb + 32 < K)
            __builtin_prefetch(&W[(size_t)(kb + 32 + (tid & 31)) * N + nb + (tid >> 5) * 16], 0, 3);
    };

    v8f acc[2][4] = {};

    const int nsteps = K >> 5;
    stage(0, 0);
    if (A_F16) wait_async();
    __syncthreads();

    for (int s = 0; s < nsteps; ++s) {
        const int cur = s & 1;
        if (s + 1 < nsteps) stage(cur ^ 1, (s + 1) * 32);

        // ---- Fragments + 8 WMMAs from buffer `cur` ----
        v16h af[2];
#pragma unroll
        for (int mt = 0; mt < 2; ++mt) {
            const half_t* arow = &sA[cur][((waveM * 2 + mt) * 16 + ll) * 32];
            af[mt] = load_frag(arow + lh * 16, arow + lh * 16 + 8);
        }
#pragma unroll
        for (int nt = 0; nt < 4; ++nt) {
            const half_t* brow = &sB[cur][((waveN * 4 + nt) * 16 + ll) * 32 + lh * 16];
            v16h bf = load_frag(brow, brow + 8);
#pragma unroll
            for (int mt = 0; mt < 2; ++mt)
                acc[mt][nt] = WMMA_F16(af[mt], bf, acc[mt][nt]);
        }

        if (A_F16) wait_async();
        __syncthreads();
    }

#pragma unroll
    for (int nt = 0; nt < 4; ++nt) {
        const int col = nb + (waveN * 4 + nt) * 16 + ll;
        const float bv = bias[col];
#pragma unroll
        for (int mt = 0; mt < 2; ++mt) {
#pragma unroll
            for (int r = 0; r < 8; ++r) {
                const int row = mb + (waveM * 2 + mt) * 16 + r + 8 * lh;
                const float v = acc[mt][nt][r] + bv;
                if (OUT_F32) ((float*)Out)[(size_t)row * N + col] = v;
                else         ((half_t*)Out)[(size_t)row * N + col] = (half_t)v;
            }
        }
    }
}

// ---------------------------------------------------------------------------
// Flash attention (causal, no 1/sqrt(d) scale per reference).
// qkv: f16 [4096][3072] (Q|K|V per token). attn_out: f16 [4096][1024].
// One workgroup per (batch, head, 128-query block); wave owns 16 query rows.
// Double-buffered K/V staging; row-sum computed as P @ ones via WMMA.
// ---------------------------------------------------------------------------
__global__ __launch_bounds__(256) void attn_kernel(
    const half_t* __restrict__ qkv, half_t* __restrict__ attn_out)
{
    constexpr int SEQ = 2048, LD = 3072, HD = 64;
    __shared__ __align__(16) half_t sQ[128 * 64];        // [qrow][d]
    __shared__ __align__(16) half_t sK[2][32 * 64];      // [key][d]
    __shared__ __align__(16) half_t sVt[2][64 * 32];     // [d][key]  (transposed)
    __shared__ __align__(16) half_t sP[8 * 16 * 32];     // per-wave P scratch

    const int tid = threadIdx.x, wave = tid >> 5, lane = tid & 31;
    const int lh = lane >> 4, ll = lane & 15;
    const int bh = blockIdx.y, b = bh >> 4, h = bh & 15;
    const int qb = blockIdx.x * 128;

    const half_t* qp = qkv + (size_t)b * SEQ * LD + h * HD;
    const half_t* kp = qp + 1024;
    const half_t* vp = qp + 2048;

    // Stage Q via async DMA: 1024 x 16B chunks, 4 per thread.
#pragma unroll
    for (int j = 0; j < 4; ++j) {
        const int chunk = tid + j * 256;
        const int r = chunk >> 3;               // 8 chunks per 64-half row
        const int d = (chunk & 7) * 8;
        async_copy_b128(qp + (size_t)(qb + r) * LD + d, &sQ[r * 64 + d]);
    }

    auto stageKV = [&](int buf, int kb) {
        // K tile: raw async DMA (256 x 16B chunks, 1 per thread).
        const int kk = tid >> 3;
        const int d8 = (tid & 7) * 8;
        async_copy_b128(kp + (size_t)(kb + kk) * LD + d8, &sK[buf][kk * 64 + d8]);
        // V tile: transposed stage (element reshuffle -> VALU path).
        for (int i = tid; i < 32 * 64; i += 256) {
            const int k2 = i >> 6, d = i & 63;
            sVt[buf][d * 32 + k2] = vp[(size_t)(kb + k2) * LD + d];
        }
    };

    wait_async();
    __syncthreads();

    // Resident Q fragments, d in [0,32) and [32,64).
    v16h qf[2];
    {
        const half_t* qrow = &sQ[(wave * 16 + ll) * 64];
#pragma unroll
        for (int s = 0; s < 2; ++s)
            qf[s] = load_frag(qrow + s * 32 + lh * 16, qrow + s * 32 + lh * 16 + 8);
    }
    // All-ones B fragment: P @ ones == per-row sum, replicated across lanes.
    v16h onesf;
#pragma unroll
    for (int i = 0; i < 16; ++i) onesf[i] = (half_t)1.0f;

    float m[8], lsum[8];
#pragma unroll
    for (int r = 0; r < 8; ++r) { m[r] = -1e30f; lsum[r] = 0.f; }
    v8f o[4] = {};

    const int nchunks = (qb + 128) >> 5;       // causal: keys 0 .. qb+127
    stageKV(0, 0);
    wait_async();
    __syncthreads();

    for (int c = 0; c < nchunks; ++c) {
        const int cur = c & 1;
        const int kb = c * 32;
        if (c + 1 < nchunks) stageKV(cur ^ 1, kb + 32);

        // S = Q K^T for two 16-key subtiles (contraction over d=64 in 2 steps).
        v8f s0 = {}, s1 = {};
#pragma unroll
        for (int t = 0; t < 2; ++t) {
            const half_t* krow = &sK[cur][(t * 16 + ll) * 64];
            v16h bk0 = load_frag(krow + lh * 16,      krow + lh * 16 + 8);
            v16h bk1 = load_frag(krow + 32 + lh * 16, krow + 32 + lh * 16 + 8);
            v8f s = {};
            s = WMMA_F16(qf[0], bk0, s);
            s = WMMA_F16(qf[1], bk1, s);
            if (t == 0) s0 = s; else s1 = s;
        }

        // Causal mask + row max (rows: lanes 0-15 -> r, lanes 16-31 -> r+8).
        const int qrow0 = qb + wave * 16 + 8 * lh;
        float rmax[8];
#pragma unroll
        for (int r = 0; r < 8; ++r) {
            const int qr = qrow0 + r;
            float a = s0[r], c2 = s1[r];
            if (kb + ll > qr)      a  = -1e30f;
            if (kb + 16 + ll > qr) c2 = -1e30f;
            s0[r] = a; s1[r] = c2;
            rmax[r] = reduce_max16(fmaxf(a, c2));
        }

        // Online softmax update (row sums come from the ones-WMMA below).
        float p0[8], p1[8];
#pragma unroll
        for (int r = 0; r < 8; ++r) {
            const float nm = fmaxf(m[r], rmax[r]);
            const float sc = __expf(m[r] - nm);
            m[r] = nm;
            p0[r] = __expf(s0[r] - nm);
            p1[r] = __expf(s1[r] - nm);
            lsum[r] *= sc;
#pragma unroll
            for (int nt = 0; nt < 4; ++nt) o[nt][r] *= sc;
        }

        // Re-layout P (C-layout -> A-fragment layout) through per-wave LDS tile.
        half_t* Pw = &sP[wave * 512];
#pragma unroll
        for (int r = 0; r < 8; ++r) {
            const int prow = r + 8 * lh;
            Pw[prow * 32 + ll]      = (half_t)p0[r];
            Pw[prow * 32 + 16 + ll] = (half_t)p1[r];
        }
        asm volatile("s_wait_dscnt 0" ::: "memory");  // wave-private tile: DS in-order + wait

        const half_t* prow = &Pw[ll * 32];
        v16h pf = load_frag(prow + lh * 16, prow + lh * 16 + 8);

        // Row-sum of P via matrix op: rs[r] = sum_k P[row,k].
        v8f rs = {};
        rs = WMMA_F16(pf, onesf, rs);
#pragma unroll
        for (int r = 0; r < 8; ++r) lsum[r] += rs[r];

#pragma unroll
        for (int nt = 0; nt < 4; ++nt) {
            const half_t* vrow = &sVt[cur][(nt * 16 + ll) * 32 + lh * 16];
            v16h vf = load_frag(vrow, vrow + 8);
            o[nt] = WMMA_F16(pf, vf, o[nt]);
        }

        wait_async();
        __syncthreads();
    }

    // Normalize and store (f16, [B*S][1024], head-interleaved columns).
#pragma unroll
    for (int r = 0; r < 8; ++r) {
        const float inv = 1.f / lsum[r];
        const int row = b * SEQ + qb + wave * 16 + r + 8 * lh;
#pragma unroll
        for (int nt = 0; nt < 4; ++nt) {
            const int col = h * HD + nt * 16 + ll;
            attn_out[(size_t)row * 1024 + col] = (half_t)(o[nt][r] * inv);
        }
    }
}

// ---------------------------------------------------------------------------
extern "C" void kernel_launch(void* const* d_in, const int* in_sizes, int n_in,
                              void* d_out, int out_size, void* d_ws, size_t ws_size,
                              hipStream_t stream) {
    const float* hs     = (const float*)d_in[0];
    const float* w_attn = (const float*)d_in[1];
    const float* b_attn = (const float*)d_in[2];
    const float* w_proj = (const float*)d_in[3];
    const float* b_proj = (const float*)d_in[4];
    float* out = (float*)d_out;

    half_t* qkv_ws  = (half_t*)d_ws;                       // [4096][3072] f16
    half_t* attn_ws = qkv_ws + (size_t)4096 * 3072;        // [4096][1024] f16

    // 1) QKV = X @ W_attn + b_attn  -> f16 workspace
    gemm_bias_kernel<false, false><<<dim3(3072 / 128, 4096 / 128), 256, 0, stream>>>(
        hs, w_attn, b_attn, qkv_ws, 4096, 3072, 1024);

    // 2) Causal flash attention over 16 heads -> f16 workspace
    attn_kernel<<<dim3(2048 / 128, 2 * 16), 256, 0, stream>>>(qkv_ws, attn_ws);

    // 3) Out = attn @ W_proj + b_proj -> f32 output
    gemm_bias_kernel<true, true><<<dim3(1024 / 128, 4096 / 128), 256, 0, stream>>>(
        attn_ws, w_proj, b_proj, out, 4096, 1024, 1024);
}
